// RGCNLayer_8693013807306
// MI455X (gfx1250) — matrix operand order repaired and checked by reading, not compiled, not run on hardware
//
#include <hip/hip_runtime.h>
#include <hip/hip_bf16.h>
#include <stdint.h>

// ---------------------------------------------------------------------------
// RGCN layer for MI455X (gfx1250, wave32, WMMA).
//  h:[N,128] f32, weight:[8,128,128], gate_w:[8,128,1], bias:[128],
//  loop_w:[128,128], norm:[E,1], src/dst/rel:[E] i32  ->  out:[N,128] f32
// ---------------------------------------------------------------------------

typedef __attribute__((ext_vector_type(16))) __bf16 v16bf;
typedef __attribute__((ext_vector_type(8)))  float  v8f;

#define IN_FEAT 128
#define OUT_FEAT 128
#define NUM_RELS 8

union FragAB {            // 16 bf16 per lane == 8 VGPRs == two b128 loads
    v16bf v;
    uint4 q[2];
};

// ---------------------------------------------------------------------------
// 1) Convert h -> bf16 and pack weights (8 rel + loop) transposed to
//    wt[rel][out][in] bf16 so B fragments are contiguous-in-K per lane.
// ---------------------------------------------------------------------------
__global__ void convert_kernel(const float* __restrict__ h,
                               const float* __restrict__ w,
                               const float* __restrict__ lw,
                               __bf16* __restrict__ hb,
                               __bf16* __restrict__ wt,
                               int n_h) {
    int stride = gridDim.x * blockDim.x;
    for (int i = blockIdx.x * blockDim.x + threadIdx.x; i < n_h; i += stride)
        hb[i] = (__bf16)h[i];
    const int n_w = (NUM_RELS + 1) * IN_FEAT * OUT_FEAT;
    for (int i = blockIdx.x * blockDim.x + threadIdx.x; i < n_w; i += stride) {
        int r = i / (IN_FEAT * OUT_FEAT);
        int rem = i - r * (IN_FEAT * OUT_FEAT);
        int o = rem >> 7;          // out index
        int k = rem & 127;         // in index (contiguous in dest)
        float v = (r < NUM_RELS) ? w[(r * IN_FEAT + k) * OUT_FEAT + o]
                                 : lw[k * OUT_FEAT + o];
        wt[i] = (__bf16)v;
    }
}

// ---------------------------------------------------------------------------
// 2) Per-(relation, node) sigmoid gate. One wave per node, 8 reductions.
// ---------------------------------------------------------------------------
__global__ __launch_bounds__(32) void gate_kernel(const float* __restrict__ h,
                                                  const float* __restrict__ gw,
                                                  float* __restrict__ gate,
                                                  int n_nodes) {
    int n = blockIdx.x;
    int l = threadIdx.x;
    float hv0 = h[n * IN_FEAT + l];
    float hv1 = h[n * IN_FEAT + l + 32];
    float hv2 = h[n * IN_FEAT + l + 64];
    float hv3 = h[n * IN_FEAT + l + 96];
#pragma unroll
    for (int r = 0; r < NUM_RELS; ++r) {
        const float* g = gw + r * IN_FEAT;
        float s = hv0 * g[l] + hv1 * g[l + 32] + hv2 * g[l + 64] + hv3 * g[l + 96];
#pragma unroll
        for (int off = 16; off > 0; off >>= 1) s += __shfl_down(s, off, 32);
        if (l == 0) gate[r * n_nodes + n] = 1.0f / (1.0f + __expf(-s));
    }
}

// ---------------------------------------------------------------------------
// 3) WMMA GEMM. 4 waves / block; each wave owns one 16-row tile and computes
//    ALL 9 transforms from one in-register copy of A (16x128 bf16 of h):
//      r=0..7 : h_rel[r] (bf16, npad-padded rows -> NO bounds checks)
//      self   : d_out = h@loop_w + bias (f32, single uniform tail branch)
//    Inner product: 8 column tiles x 4 K-steps of v_wmma_f32_16x16x32_bf16.
// ---------------------------------------------------------------------------
__global__ __launch_bounds__(128) void gemm_kernel(const __bf16* __restrict__ hb,
                                                   const __bf16* __restrict__ wt,
                                                   const float* __restrict__ bias,
                                                   __bf16* __restrict__ hrel,
                                                   float* __restrict__ out,
                                                   int n_nodes, int npad) {
    const int ntiles = npad >> 4;
    const int tile   = blockIdx.x * 4 + (threadIdx.x >> 5);
    if (tile >= ntiles) return;
    const int row0 = tile * 16;
    const int l    = threadIdx.x & 31;
    const int m    = l & 15;        // A row / B col within tile
    const int half = l >> 4;

    // --- A fragments: ISA 16-bit A 16x32 layout.
    // lane(m,half): VGPR0-3 hold K = ks*32 + 8*half + [0..7]   (contiguous)
    //               VGPR4-7 hold K = ks*32 + 16 + 8*half + [0..7]
    int arow = row0 + m;
    if (arow >= n_nodes) arow = n_nodes - 1;   // read clamp only
    const __bf16* hrow = hb + (size_t)arow * IN_FEAT;
    FragAB A[4];
#pragma unroll
    for (int ks = 0; ks < 4; ++ks) {
        A[ks].q[0] = *(const uint4*)(hrow + ks * 32 + half * 8);
        A[ks].q[1] = *(const uint4*)(hrow + ks * 32 + 16 + half * 8);
    }

    // ---- relations 0..7 -> padded h_rel, guard-free stores -----------------
#pragma unroll 1
    for (int r = 0; r < NUM_RELS; ++r) {
        const __bf16* wrel = wt + (size_t)r * IN_FEAT * OUT_FEAT;   // [out][in]
#pragma unroll
        for (int nt = 0; nt < 8; ++nt) {
            v8f c = {};
#pragma unroll
            for (int ks = 0; ks < 4; ++ks) {
                // B fragment: lane(n=m,half): K = ks*32 + 8*half + [0..7], ks*32 + 16 + 8*half + [0..7]
                FragAB B;
                const __bf16* bp = wrel + (size_t)(nt * 16 + m) * IN_FEAT
                                        + ks * 32 + half * 8;
                B.q[0] = *(const uint4*)(bp);
                B.q[1] = *(const uint4*)(bp + 16);
                c = __builtin_amdgcn_wmma_f32_16x16x32_bf16(
                        false, A[ks].v, false, B.v, (short)0, c, false, false);
            }
            // C/D layout: lane(n=m,half), VGPR j -> row m' = j + 8*half
            const int n = nt * 16 + m;
            __bf16* dp = hrel + ((size_t)r * npad + row0 + 8 * half) * OUT_FEAT + n;
#pragma unroll
            for (int j = 0; j < 8; ++j)
                dp[(size_t)j * OUT_FEAT] = (__bf16)c[j];
        }
    }

    // ---- self-loop -> d_out (+bias), single uniform tail branch ------------
    {
        const __bf16* wrel = wt + (size_t)NUM_RELS * IN_FEAT * OUT_FEAT;
        const bool full = (row0 + 16) <= n_nodes;   // tile-uniform
#pragma unroll
        for (int nt = 0; nt < 8; ++nt) {
            v8f c = {};
#pragma unroll
            for (int ks = 0; ks < 4; ++ks) {
                FragAB B;
                const __bf16* bp = wrel + (size_t)(nt * 16 + m) * IN_FEAT
                                        + ks * 32 + half * 8;
                B.q[0] = *(const uint4*)(bp);
                B.q[1] = *(const uint4*)(bp + 16);
                c = __builtin_amdgcn_wmma_f32_16x16x32_bf16(
                        false, A[ks].v, false, B.v, (short)0, c, false, false);
            }
            const int n  = nt * 16 + m;
            float bv     = bias[n];
            float* op    = out + (size_t)(row0 + 8 * half) * OUT_FEAT + n;
            if (full) {
#pragma unroll
                for (int j = 0; j < 8; ++j)
                    op[(size_t)j * OUT_FEAT] = c[j] + bv;
            } else {
#pragma unroll
                for (int j = 0; j < 8; ++j)
                    if (row0 + 8 * half + j < n_nodes)
                        op[(size_t)j * OUT_FEAT] = c[j] + bv;
            }
        }
    }
}

// ---------------------------------------------------------------------------
// 4) Edge kernel: one wave per edge. Gather bf16 h_rel row, scale by
//    norm * gate, atomic scatter-add into destination row (f32).
// ---------------------------------------------------------------------------
__global__ __launch_bounds__(256) void edge_kernel(const __bf16* __restrict__ hrel,
                                                   const float* __restrict__ norm,
                                                   const int* __restrict__ src,
                                                   const int* __restrict__ dst,
                                                   const int* __restrict__ rel,
                                                   const float* __restrict__ gate,
                                                   float* __restrict__ out,
                                                   int n_edges, int n_nodes,
                                                   int npad) {
    int e = (blockIdx.x * blockDim.x + threadIdx.x) >> 5;
    int l = threadIdx.x & 31;
    if (e >= n_edges) return;
    int s = src[e];
    int d = dst[e];
    int r = rel[e];
    float coeff = norm[e] * gate[(size_t)r * n_nodes + s];
    const __bf16* row = hrel + ((size_t)r * npad + s) * OUT_FEAT;
    union { uint2 u; __bf16 b[4]; } pk;
    pk.u = *(const uint2*)(row + l * 4);
    float* o = out + (size_t)d * OUT_FEAT + l * 4;
#pragma unroll
    for (int j = 0; j < 4; ++j)
        atomicAdd(o + j, (float)pk.b[j] * coeff);
}

// ---------------------------------------------------------------------------
// 5) In-place ReLU on d_out.
// ---------------------------------------------------------------------------
__global__ void relu_kernel(float* __restrict__ out, int n) {
    int stride = gridDim.x * blockDim.x;
    for (int i = blockIdx.x * blockDim.x + threadIdx.x; i < n; i += stride)
        out[i] = fmaxf(out[i], 0.0f);
}

// ---------------------------------------------------------------------------
extern "C" void kernel_launch(void* const* d_in, const int* in_sizes, int n_in,
                              void* d_out, int out_size, void* d_ws, size_t ws_size,
                              hipStream_t stream) {
    const float* h    = (const float*)d_in[0];
    const float* w    = (const float*)d_in[1];
    const float* gw   = (const float*)d_in[2];
    const float* bias = (const float*)d_in[3];
    const float* lw   = (const float*)d_in[4];
    const float* norm = (const float*)d_in[5];
    const int*   src  = (const int*)d_in[6];
    const int*   dst  = (const int*)d_in[7];
    const int*   rel  = (const int*)d_in[8];
    float* out = (float*)d_out;

    const int n_nodes = in_sizes[0] / IN_FEAT;
    const int n_edges = in_sizes[5];
    const int n_h     = n_nodes * IN_FEAT;
    const int ntiles  = (n_nodes + 15) / 16;
    const int npad    = ntiles * 16;           // padded node count for h_rel

    // Workspace layout (bytes from d_ws base, all 16B-aligned):
    //  hb   : n_h bf16                  (12.8 MB)
    //  wt   : 9*128*128 bf16            (0.29 MB)
    //  gate : 8*n_nodes f32             (1.6 MB)
    //  hrel : 8*npad*128 bf16 (padded)  (102.4 MB)
    __bf16* hb   = (__bf16*)d_ws;
    __bf16* wt   = hb + (size_t)n_h;
    size_t  goff = ((size_t)n_h + (NUM_RELS + 1) * IN_FEAT * OUT_FEAT) * sizeof(__bf16);
    goff = (goff + 15) & ~(size_t)15;
    float*  gate = (float*)((char*)d_ws + goff);
    size_t  hoff = goff + (size_t)NUM_RELS * n_nodes * sizeof(float);
    hoff = (hoff + 15) & ~(size_t)15;
    __bf16* hrel = (__bf16*)((char*)d_ws + hoff);

    // 1) precision conversion + weight transpose
    convert_kernel<<<(n_h + 255) / 256, 256, 0, stream>>>(h, w, lw, hb, wt, n_h);

    // 2) gates
    gate_kernel<<<n_nodes, 32, 0, stream>>>(h, gw, gate, n_nodes);

    // 3) all 9 GEMMs via WMMA, one A-load per 16-row tile
    gemm_kernel<<<(ntiles + 3) / 4, 128, 0, stream>>>(hb, wt, bias, hrel, out,
                                                      n_nodes, npad);

    // 4) gather / scale / scatter-add
    int eblocks = (n_edges * 32 + 255) / 256;
    edge_kernel<<<eblocks, 256, 0, stream>>>(hrel, norm, src, dst, rel, gate,
                                             out, n_edges, n_nodes, npad);

    // 5) activation
    relu_kernel<<<(n_h + 255) / 256, 256, 0, stream>>>(out, n_h);
}
